// DiffPoolGNN_74938589380725
// MI455X (gfx1250) — matrix operand order repaired and checked by reading, not compiled, not run on hardware
//
#include <hip/hip_runtime.h>
#include <hip/hip_bf16.h>

typedef __attribute__((ext_vector_type(16))) _Float16 v16h;
typedef __attribute__((ext_vector_type(8)))  _Float16 v8h;
typedef __attribute__((ext_vector_type(8)))  float    v8f;

// ---------------- problem constants (match reference) ----------------
static constexpr int BG    = 64;
static constexpr int NPER  = 512;
static constexpr int DEG   = 32;
static constexpr int NTOT  = BG * NPER;          // 32768
static constexpr int EDGES = NTOT * DEG;         // 1048576
static constexpr int NTILES = EDGES / 16;        // 65536
static constexpr int HID   = 64;
static constexpr int IN_D  = 16;
static constexpr int ED_D  = 8;
static constexpr int AC    = 16;                 // assign clusters
static constexpr int EMB   = 128;

// ---------------- wave-local LDS sync (single-wave producer/consumer) ----
__device__ inline void wave_lds_sync() {
  __builtin_amdgcn_wave_barrier();
  asm volatile("s_wait_dscnt 0" ::: "memory");
  __builtin_amdgcn_wave_barrier();
}

// ---------------- utility kernels ----------------
__global__ void zero_kernel(float* __restrict__ p, int n) {
  for (int i = blockIdx.x * blockDim.x + threadIdx.x; i < n; i += gridDim.x * blockDim.x)
    p[i] = 0.0f;
}

__global__ void cvt_f16_kernel(const float* __restrict__ src, _Float16* __restrict__ dst, int n) {
  for (int i = blockIdx.x * blockDim.x + threadIdx.x; i < n; i += gridDim.x * blockDim.x)
    dst[i] = (_Float16)src[i];
}

__global__ void count_kernel(const int* __restrict__ dst, float* __restrict__ cnt) {
  for (int e = blockIdx.x * blockDim.x + threadIdx.x; e < EDGES; e += gridDim.x * blockDim.x)
    atomicAdd(&cnt[dst[e]], 1.0f);
}

// h = relu(msum / max(cnt,1)) stored as f16
__global__ void mean_relu_kernel(const float* __restrict__ msum, const float* __restrict__ cnt,
                                 _Float16* __restrict__ h, int total, int outd) {
  int i = blockIdx.x * blockDim.x + threadIdx.x;
  if (i >= total) return;
  int n = i / outd;
  h[i] = (_Float16)fmaxf(msum[i] / fmaxf(cnt[n], 1.0f), 0.0f);
}

__global__ void softmax_s_kernel(const float* __restrict__ ssum, const float* __restrict__ cnt,
                                 _Float16* __restrict__ s) {
  int n = blockIdx.x * blockDim.x + threadIdx.x;
  if (n >= NTOT) return;
  float c = fmaxf(cnt[n], 1.0f);
  float v[AC];
  float mx = -1e30f;
  #pragma unroll
  for (int i = 0; i < AC; ++i) { v[i] = ssum[n * AC + i] / c; mx = fmaxf(mx, v[i]); }
  float sum = 0.0f;
  #pragma unroll
  for (int i = 0; i < AC; ++i) { v[i] = expf(v[i] - mx); sum += v[i]; }
  float inv = 1.0f / sum;
  #pragma unroll
  for (int i = 0; i < AC; ++i) s[n * AC + i] = (_Float16)(v[i] * inv);
}

// ---------------- WMMA fragment index helpers (per ISA VGPR layout) -------
__device__ inline int a_kk(int hi, int i) {           // 16-bit A 16x32
  return ((i < 8) ? i : (i + 8)) + (hi ? 8 : 0);
}
__device__ inline int b_kk(int hi, int i) {           // 16-bit B 32x16
  return ((i < 8) ? i : (i + 8)) + (hi ? 8 : 0);
}

// 8 contiguous f16 of an A-row group starting at K=kk0.
// Row layout: [0,DN) node feats | [DN,DN+8) edge attr | [DN+8,K1PAD) zero.
// DN and DN+8 are multiples of 8 so each group lies wholly in one region.
template<int DN>
__device__ inline v8h grp8(const _Float16* __restrict__ nfrow,
                           const _Float16* __restrict__ earow, int kk0) {
  if (kk0 + 8 <= DN) return *(const v8h*)(nfrow + kk0);
  if (kk0 >= DN && kk0 < DN + ED_D) return *(const v8h*)earow;
  v8h z = {};
  return z;
}

// ---------------- fused edge-MLP conv:  mean_dst( W2*relu(W1*[x[src];ea]+b1)+b2 )
template<int DN, int OUT>
__global__ __launch_bounds__(256) void edge_conv_wmma(
    const _Float16* __restrict__ nf, const int* __restrict__ srcI, const int* __restrict__ dstI,
    const _Float16* __restrict__ eah,
    const float* __restrict__ w1, const float* __restrict__ b1,
    const float* __restrict__ w2, const float* __restrict__ b2,
    float* __restrict__ msum, int ntiles)
{
  constexpr int K1PAD = ((DN + ED_D + 31) / 32) * 32;
  constexpr int K2PAD = (OUT < 32) ? 32 : OUT;
  constexpr int NC  = OUT / 16;
  constexpr int KC1 = K1PAD / 32;
  constexpr int KC2 = K2PAD / 32;

  // weights pre-swizzled into WMMA B-fragment order: one v16h per (chunk, lane)
  __shared__ v16h    sW1f[KC1 * NC * 32];
  __shared__ v16h    sW2f[KC2 * NC * 32];
  __shared__ float   sB1[OUT], sB2[OUT];
  __shared__ _Float16 sM[8][16 * K2PAD];     // per-wave mid tile (layer1 out)

  const int tid = threadIdx.x;
  // swizzle W1: element i of fragment (kc,nc,lane) <- w1[kc*32 + b_kk(hi,i)][nc*16 + r]
  for (int idx = tid; idx < KC1 * NC * 32 * 16; idx += 256) {
    const int i  = idx & 15;
    const int ln = (idx >> 4) & 31;
    const int cn = idx >> 9;
    const int kc = cn / NC, nc = cn % NC;
    const int k  = kc * 32 + b_kk(ln >> 4, i);
    const int n  = nc * 16 + (ln & 15);
    ((_Float16*)sW1f)[idx] = (k < DN + ED_D) ? (_Float16)w1[k * OUT + n] : (_Float16)0.0f;
  }
  for (int idx = tid; idx < KC2 * NC * 32 * 16; idx += 256) {
    const int i  = idx & 15;
    const int ln = (idx >> 4) & 31;
    const int cn = idx >> 9;
    const int kc = cn / NC, nc = cn % NC;
    const int k  = kc * 32 + b_kk(ln >> 4, i);
    const int n  = nc * 16 + (ln & 15);
    ((_Float16*)sW2f)[idx] = (k < OUT) ? (_Float16)w2[k * OUT + n] : (_Float16)0.0f;
  }
  for (int i = tid; i < OUT; i += 256) { sB1[i] = b1[i]; sB2[i] = b2[i]; }

  const int wave = tid >> 5, lane = tid & 31;
  const int r = lane & 15, hi = lane >> 4;
  _Float16* M = sM[wave];
  for (int i = lane; i < 16 * K2PAD; i += 32) M[i] = (_Float16)0.0f;  // zero K2 pad once
  __syncthreads();

  const int gw = blockIdx.x * 8 + wave;
  const int nw = gridDim.x * 8;

  for (int t = gw; t < ntiles; t += nw) {
    const int base = t * 16;
    const int e    = base + r;
    const int sp   = srcI[e];
    const _Float16* nfrow = nf + (size_t)sp * DN;
    const _Float16* earow = eah + (size_t)e * ED_D;

    // prefetch next tile's gather sources
    {
      const int tn = t + nw;
      if (tn < ntiles) {
        __builtin_prefetch(srcI + tn * 16, 0, 1);
        __builtin_prefetch(dstI + tn * 16, 0, 1);
        __builtin_prefetch(eah + (size_t)tn * 16 * ED_D, 0, 1);
      }
    }

    // ---- layer 1: [16 x K1] @ [K1 x OUT]; A built directly in registers ----
    v8f acc[NC] = {};
    #pragma unroll
    for (int kc = 0; kc < KC1; ++kc) {
      const int kb = kc * 32;
      const v8h lo = grp8<DN>(nfrow, earow, kb + (hi ? 8 : 0));
      const v8h up = grp8<DN>(nfrow, earow, kb + 16 + (hi ? 8 : 0));
      const v16h afr = __builtin_shufflevector(lo, up,
          0, 1, 2, 3, 4, 5, 6, 7, 8, 9, 10, 11, 12, 13, 14, 15);
      #pragma unroll
      for (int nc = 0; nc < NC; ++nc) {
        const v16h bfr = sW1f[(kc * NC + nc) * 32 + lane];
        acc[nc] = __builtin_amdgcn_wmma_f32_16x16x32_f16(false, afr, false, bfr,
                                                         (short)0, acc[nc], false, false);
      }
    }

    // ---- bias + relu -> mid tile (f16, row-major) ----
    #pragma unroll
    for (int nc = 0; nc < NC; ++nc) {
      const int col = nc * 16 + r;
      const float bb = sB1[col];
      #pragma unroll
      for (int j = 0; j < 8; ++j) {
        const int row = j + (hi ? 8 : 0);
        M[row * K2PAD + col] = (_Float16)fmaxf(acc[nc][j] + bb, 0.0f);
      }
    }
    wave_lds_sync();

    // ---- layer 2: [16 x OUT] @ [OUT x OUT]; A fragments = 2 wide LDS loads ----
    v8f acc2[NC] = {};
    #pragma unroll
    for (int kc = 0; kc < KC2; ++kc) {
      const int kb = kc * 32;
      const v8h lo = *(const v8h*)(M + r * K2PAD + kb + (hi ? 8 : 0));
      const v8h up = *(const v8h*)(M + r * K2PAD + kb + 16 + (hi ? 8 : 0));
      const v16h afr = __builtin_shufflevector(lo, up,
          0, 1, 2, 3, 4, 5, 6, 7, 8, 9, 10, 11, 12, 13, 14, 15);
      #pragma unroll
      for (int nc = 0; nc < NC; ++nc) {
        const v16h bfr = sW2f[(kc * NC + nc) * 32 + lane];
        acc2[nc] = __builtin_amdgcn_wmma_f32_16x16x32_f16(false, afr, false, bfr,
                                                          (short)0, acc2[nc], false, false);
      }
    }

    // ---- bias + scatter-add to dst accumulators ----
    const int* dp = dstI + base + (hi ? 8 : 0);
    #pragma unroll
    for (int nc = 0; nc < NC; ++nc) {
      const int col = nc * 16 + r;
      const float bb = sB2[col];
      #pragma unroll
      for (int j = 0; j < 8; ++j) {
        atomicAdd(&msum[(size_t)dp[j] * OUT + col], acc2[nc][j] + bb);
      }
    }
    wave_lds_sync();   // mid tile reuse across iterations
  }
}

// ---------------- x1[b,c,f] = sum_n s[b,n,c] * h[b,n,f]  (M=16,K=512,N=64) ----
// Rows of the K-chunk are staged into LDS with wide coalesced copies; the
// WMMA fragments are then built with constant-offset ds loads.
__global__ __launch_bounds__(128) void pool_x1_kernel(const _Float16* __restrict__ s,
                                                      const _Float16* __restrict__ h,
                                                      float* __restrict__ x1) {
  __shared__ _Float16 sS[32 * AC];     // 32 node rows of s  (1 KB)
  __shared__ _Float16 sH[32 * HID];    // 32 node rows of h  (4 KB)
  const int b = blockIdx.x;
  const int tid = threadIdx.x;
  const int wave = tid >> 5, lane = tid & 31;
  const int r = lane & 15, hi = lane >> 4;
  const int f0 = wave * 16;
  v8f acc = {};
  for (int kc = 0; kc < NPER / 32; ++kc) {
    const int k0 = kc * 32;
    if (tid < 64) {                               // s: row = tid>>1, half = tid&1
      const int row = tid >> 1, hf = tid & 1;
      *(v8h*)(sS + row * AC + hf * 8) =
          *(const v8h*)(s + (size_t)(b * NPER + k0 + row) * AC + hf * 8);
    }
    {                                             // h: row = tid>>2, quarter = tid&3
      const int row = tid >> 2, q = tid & 3;
      const _Float16* hp = h + (size_t)(b * NPER + k0 + row) * HID + q * 16;
      *(v8h*)(sH + row * HID + q * 16)     = *(const v8h*)(hp);
      *(v8h*)(sH + row * HID + q * 16 + 8) = *(const v8h*)(hp + 8);
    }
    __syncthreads();
    v16h afr, bfr;
    #pragma unroll
    for (int i = 0; i < 16; ++i) {
      afr[i] = sS[a_kk(hi, i) * AC + r];
      bfr[i] = sH[b_kk(hi, i) * HID + f0 + r];
    }
    acc = __builtin_amdgcn_wmma_f32_16x16x32_f16(false, afr, false, bfr,
                                                 (short)0, acc, false, false);
    __syncthreads();
  }
  #pragma unroll
  for (int j = 0; j < 8; ++j) {
    const int row = j + (hi ? 8 : 0);
    x1[(size_t)(b * AC + row) * HID + f0 + r] = acc[j];
  }
}

// ---------------- adj1[b,c,k] = sum_edges s[src,c]*s[dst,k]  (M=N=16, K=16384) ----
__global__ __launch_bounds__(256) void adj1_kernel(const _Float16* __restrict__ s,
                                                   const int* __restrict__ srcI,
                                                   const int* __restrict__ dstI,
                                                   float* __restrict__ adj1) {
  __shared__ float part[8][256];
  __shared__ _Float16 rowsS[8][32 * AC];   // per-wave staged s[src] rows (1 KB each)
  __shared__ _Float16 rowsD[8][32 * AC];   // per-wave staged s[dst] rows
  const int b = blockIdx.x;
  const int wave = threadIdx.x >> 5, lane = threadIdx.x & 31;
  const int r = lane & 15, hi = lane >> 4;
  const int EPG = NPER * DEG;                 // 16384 edges per graph
  _Float16* rs = rowsS[wave];
  _Float16* rd = rowsD[wave];
  v8f acc = {};
  for (int kc = wave; kc < EPG / 32; kc += 8) {
    const int k0 = b * EPG + kc * 32;
    const int se = srcI[k0 + lane];
    const int de = dstI[k0 + lane];
    const _Float16* sp = s + (size_t)se * AC;
    const _Float16* dq = s + (size_t)de * AC;
    *(v8h*)(rs + lane * AC)     = *(const v8h*)(sp);
    *(v8h*)(rs + lane * AC + 8) = *(const v8h*)(sp + 8);
    *(v8h*)(rd + lane * AC)     = *(const v8h*)(dq);
    *(v8h*)(rd + lane * AC + 8) = *(const v8h*)(dq + 8);
    wave_lds_sync();
    v16h afr, bfr;
    #pragma unroll
    for (int i = 0; i < 16; ++i) {
      afr[i] = rs[a_kk(hi, i) * AC + r];
      bfr[i] = rd[b_kk(hi, i) * AC + r];
    }
    acc = __builtin_amdgcn_wmma_f32_16x16x32_f16(false, afr, false, bfr,
                                                 (short)0, acc, false, false);
    wave_lds_sync();
  }
  #pragma unroll
  for (int j = 0; j < 8; ++j) {
    const int row = j + (hi ? 8 : 0);
    part[wave][row * 16 + r] = acc[j];
  }
  __syncthreads();
  const int t = threadIdx.x;
  float v = 0.0f;
  #pragma unroll
  for (int w = 0; w < 8; ++w) v += part[w][t];
  adj1[b * 256 + t] = v;
}

// ---------------- tail: dense GCN x2, cluster softmax-pool, final MLP -> out[b] ----
__global__ __launch_bounds__(64) void tail_kernel(
    const float* __restrict__ x1, const float* __restrict__ adj1g,
    const float* __restrict__ d1w, const float* __restrict__ d1b,
    const float* __restrict__ d2w, const float* __restrict__ d2b,
    const float* __restrict__ adw, const float* __restrict__ adb,
    const float* __restrict__ mw1, const float* __restrict__ mb1,
    const float* __restrict__ mw2, const float* __restrict__ mb2,
    float* __restrict__ out)
{
  __shared__ float adjn[256];
  __shared__ float xL[AC * HID];
  __shared__ float yL[AC * HID];
  __shared__ float tmp[48];
  const int b = blockIdx.x, f = threadIdx.x;

  for (int i = f; i < 256; i += 64) {
    float v = adj1g[b * 256 + i];
    if ((i >> 4) == (i & 15)) v = 1.0f;
    adjn[i] = v;
  }
  __syncthreads();
  if (f < AC) {
    float sum = 0.0f;
    #pragma unroll
    for (int j = 0; j < AC; ++j) sum += adjn[f * 16 + j];
    tmp[f] = rsqrtf(fmaxf(sum, 1.0f));
  }
  __syncthreads();
  for (int i = f; i < 256; i += 64) adjn[i] = tmp[i >> 4] * adjn[i] * tmp[i & 15];
  for (int i = f; i < AC * HID; i += 64) xL[i] = x1[b * AC * HID + i];
  __syncthreads();

  const float* wsel[2] = {d1w, d2w};
  const float* bsel[2] = {d1b, d2b};
  for (int L = 0; L < 2; ++L) {
    const float* w = wsel[L];
    const float bb = bsel[L][f];
    for (int m = 0; m < AC; ++m) {
      float acc = bb;
      for (int k = 0; k < HID; ++k) acc += xL[m * HID + k] * w[k * HID + f];
      yL[m * HID + f] = acc;
    }
    __syncthreads();
    for (int c = 0; c < AC; ++c) {
      float acc = 0.0f;
      #pragma unroll
      for (int m = 0; m < AC; ++m) acc += adjn[c * 16 + m] * yL[m * HID + f];
      xL[c * HID + f] = fmaxf(acc, 0.0f);
    }
    __syncthreads();
  }

  if (f < AC) {
    float z = adb[0];
    for (int k = 0; k < HID; ++k) z += xL[f * HID + k] * adw[k];
    tmp[16 + f] = z;
  }
  __syncthreads();
  if (f == 0) {
    float mx = -1e30f;
    for (int i = 0; i < AC; ++i) mx = fmaxf(mx, tmp[16 + i]);
    float sm = 0.0f;
    for (int i = 0; i < AC; ++i) { float e2 = expf(tmp[16 + i] - mx); tmp[32 + i] = e2; sm += e2; }
    float inv = 1.0f / sm;
    for (int i = 0; i < AC; ++i) tmp[32 + i] *= inv;
  }
  __syncthreads();

  float g = 0.0f;
  #pragma unroll
  for (int c = 0; c < AC; ++c) g += tmp[32 + c] * xL[c * HID + f];
  yL[f] = g;
  __syncthreads();

  for (int jj = 0; jj < 2; ++jj) {
    const int j = f + jj * 64;
    float acc = mb1[j];
    for (int k = 0; k < HID; ++k) acc += yL[k] * mw1[k * EMB + j];
    xL[j] = fmaxf(acc, 0.0f);
  }
  __syncthreads();
  if (f == 0) {
    float o = mb2[0];
    for (int j = 0; j < EMB; ++j) o += xL[j] * mw2[j];
    out[b] = o;
  }
}

// ---------------- host driver ----------------
extern "C" void kernel_launch(void* const* d_in, const int* in_sizes, int n_in,
                              void* d_out, int out_size, void* d_ws, size_t ws_size,
                              hipStream_t stream) {
  (void)in_sizes; (void)n_in; (void)out_size; (void)ws_size;
  const float* x    = (const float*)d_in[0];
  const int*  eidx  = (const int*)d_in[1];
  const float* ea   = (const float*)d_in[2];
  const int*  srcI  = eidx;
  const int*  dstI  = eidx + EDGES;
  const float* g1w1 = (const float*)d_in[4];  const float* g1b1 = (const float*)d_in[5];
  const float* g1w2 = (const float*)d_in[6];  const float* g1b2 = (const float*)d_in[7];
  const float* g2w1 = (const float*)d_in[8];  const float* g2b1 = (const float*)d_in[9];
  const float* g2w2 = (const float*)d_in[10]; const float* g2b2 = (const float*)d_in[11];
  const float* agw1 = (const float*)d_in[12]; const float* agb1 = (const float*)d_in[13];
  const float* agw2 = (const float*)d_in[14]; const float* agb2 = (const float*)d_in[15];
  const float* d1w  = (const float*)d_in[16]; const float* d1b  = (const float*)d_in[17];
  const float* d2w  = (const float*)d_in[18]; const float* d2b  = (const float*)d_in[19];
  const float* adw  = (const float*)d_in[20]; const float* adb  = (const float*)d_in[21];
  const float* mw1  = (const float*)d_in[22]; const float* mb1  = (const float*)d_in[23];
  const float* mw2  = (const float*)d_in[24]; const float* mb2  = (const float*)d_in[25];
  float* outp = (float*)d_out;

  // ---- workspace layout: f32 region then f16 region (16B aligned) ----
  float* wsf  = (float*)d_ws;
  float* cnt  = wsf;                               // 32768
  float* msum = cnt  + NTOT;                       // 32768*64
  float* ssum = msum + NTOT * HID;                 // 32768*16
  float* x1   = ssum + NTOT * AC;                  // 64*16*64
  float* adj1 = x1   + BG * AC * HID;              // 64*16*16
  _Float16* f16base = (_Float16*)(adj1 + BG * AC * AC);
  _Float16* xh  = f16base;                         // 32768*16
  _Float16* eah = xh  + NTOT * IN_D;               // 1048576*8
  _Float16* h1  = eah + (size_t)EDGES * ED_D;      // 32768*64
  _Float16* h2  = h1  + NTOT * HID;                // 32768*64
  _Float16* sh  = h2  + NTOT * HID;                // 32768*16

  // 0) convert inputs to f16
  cvt_f16_kernel<<<1024, 256, 0, stream>>>(x, xh, NTOT * IN_D);
  cvt_f16_kernel<<<4096, 256, 0, stream>>>(ea, eah, EDGES * ED_D);
  // 1) zero cnt + msum + ssum (contiguous)
  zero_kernel<<<2048, 256, 0, stream>>>(cnt, NTOT + NTOT * HID + NTOT * AC);
  // 2) per-dst edge counts
  count_kernel<<<2048, 256, 0, stream>>>(dstI, cnt);
  // 3) conv1: xh -> msum
  edge_conv_wmma<IN_D, HID><<<1024, 256, 0, stream>>>(xh, srcI, dstI, eah,
      g1w1, g1b1, g1w2, g1b2, msum, NTILES);
  // 4) h1 = relu(mean) (f16)
  mean_relu_kernel<<<(NTOT * HID + 255) / 256, 256, 0, stream>>>(msum, cnt, h1, NTOT * HID, HID);
  // 5) re-zero msum
  zero_kernel<<<2048, 256, 0, stream>>>(msum, NTOT * HID);
  // 6) conv2: h1 -> msum
  edge_conv_wmma<HID, HID><<<1024, 256, 0, stream>>>(h1, srcI, dstI, eah,
      g2w1, g2b1, g2w2, g2b2, msum, NTILES);
  // 7) h2 = relu(mean) (f16)
  mean_relu_kernel<<<(NTOT * HID + 255) / 256, 256, 0, stream>>>(msum, cnt, h2, NTOT * HID, HID);
  // 8) assign conv: h2 -> ssum
  edge_conv_wmma<HID, AC><<<1024, 256, 0, stream>>>(h2, srcI, dstI, eah,
      agw1, agb1, agw2, agb2, ssum, NTILES);
  // 9) s = softmax(mean) per node over 16 clusters (f16)
  softmax_s_kernel<<<(NTOT + 255) / 256, 256, 0, stream>>>(ssum, cnt, sh);
  // 10) x1 = s^T h2 per graph
  pool_x1_kernel<<<BG, 128, 0, stream>>>(sh, h2, x1);
  // 11) adj1 = s^T adj s via per-edge outer products
  adj1_kernel<<<BG, 256, 0, stream>>>(sh, srcI, dstI, adj1);
  // 12) dense-GCN tail + final MLP -> out[64]
  tail_kernel<<<BG, 64, 0, stream>>>(x1, adj1, d1w, d1b, d2w, d2b,
                                     adw, adb, mw1, mb1, mw2, mb2, outp);
}
